// MultiHeadAttention_80900003987928
// MI455X (gfx1250) — matrix operand order repaired and checked by reading, not compiled, not run on hardware
//
#include <hip/hip_runtime.h>

#define TSEQ   2048
#define DMODEL 2048
#define NQH    16
#define NKH    8
#define HDIM   128
#define BATCH  4

typedef __attribute__((ext_vector_type(16))) __bf16 v16bf;
typedef __attribute__((ext_vector_type(8)))  float  v8f;
typedef unsigned int tdm_v4u __attribute__((ext_vector_type(4)));
typedef int          tdm_v4i __attribute__((ext_vector_type(4)));
typedef int          tdm_v8i __attribute__((ext_vector_type(8)));

union FragAB { v16bf v; uint4 u[2]; unsigned short s[16]; };
union FragC  { v8f v; float f[8]; };

__device__ inline unsigned short f2bf(float x) {
  union { float f; unsigned int u; } c; c.f = x;
  unsigned int r = c.u + 0x7FFFu + ((c.u >> 16) & 1u);   // round-to-nearest-even
  return (unsigned short)(r >> 16);
}

__device__ inline v8f wmma_bf16(const FragAB& a, const FragAB& b, v8f c) {
  return __builtin_amdgcn_wmma_f32_16x16x32_bf16(false, a.v, false, b.v,
                                                 (short)0, c, false, false);
}

// LDS byte offset from a generic shared pointer: AMDGCN shared aperture is
// {SHARED_BASE[63:32], wave-relative offset[31:0]}, so truncation is exact.
__device__ inline unsigned lds_offset(const void* p) {
  return (unsigned)(unsigned long long)p;
}

// Async global->LDS copy, 16B per lane (VGLOBAL GLOBAL_LOAD_ASYNC_TO_LDS_B128,
// tracked by ASYNCcnt).
__device__ inline void async_copy_b128(unsigned lds_off, const void* gptr) {
  unsigned long long ga = (unsigned long long)gptr;
  asm volatile("global_load_async_to_lds_b128 %0, %1, off"
               :: "v"(lds_off), "v"(ga) : "memory");
}
__device__ inline void async_wait0() {
  asm volatile("s_wait_asynccnt 0" ::: "memory");
}

// TDM: issue a 1D/2D bf16 tensor_load_to_lds (one descriptor, whole wave).
// D# layout per CDNA5 ISA §8: group0 = {flags, lds_addr, global_addr, type=2},
// group1 = {data_size=2B, tensor dims/strides, tile dims}.
__device__ inline void tdm_load_bf16(unsigned lds_off, const void* gptr,
                                     unsigned tensor_d0, unsigned tensor_d1,
                                     unsigned tile_d0, unsigned tile_d1,
                                     unsigned stride0, unsigned stride1) {
  unsigned long long ga = (unsigned long long)gptr;
  tdm_v4u g0;
  g0.x = 1u;                                           // count=1, user mode
  g0.y = lds_off;                                      // lds_addr (bytes)
  g0.z = (unsigned)ga;                                 // global_addr[31:0]
  g0.w = (unsigned)((ga >> 32) & 0x01FFFFFFull) | 0x80000000u; // addr[56:32]|type=2
  tdm_v8i g1;
  g1[0] = 0x10000;                                     // data_size=1 -> 2 bytes
  g1[1] = (int)((tensor_d0 & 0xFFFFu) << 16);          // tensor_dim0 lo
  g1[2] = (int)(((tensor_d0 >> 16) & 0xFFFFu) | ((tensor_d1 & 0xFFFFu) << 16));
  g1[3] = (int)(((tensor_d1 >> 16) & 0xFFFFu) | ((tile_d0 & 0xFFFFu) << 16));
  g1[4] = (int)(tile_d1 & 0xFFFFu);                    // tile_dim1, tile_dim2=0
  g1[5] = (int)stride0;                                // tensor_dim0_stride lo32
  g1[6] = (int)((stride1 & 0xFFFFu) << 16);            // stride0 hi=0 | stride1 lo16
  g1[7] = (int)(stride1 >> 16);                        // stride1 hi
  tdm_v4i z4 = {0, 0, 0, 0};
#if __has_include(<hip/amd_detail/amd_gfx1250_TDM.h>)
  tdm_v8i z8 = {0, 0, 0, 0, 0, 0, 0, 0};
  __builtin_amdgcn_tensor_load_to_lds(g0, g1, z4, z4, z8, 0);
#else
  __builtin_amdgcn_tensor_load_to_lds(g0, g1, z4, z4, 0);
#endif
}

// ---------------------------------------------------------------------------
// Kernel A: QKV projection + RMSNorm + RoPE, fp32 in -> bf16 out
// grid (T/16, NQ + 2*NK, B), 256 threads (8 waves, each owns 16 H-columns)
// ---------------------------------------------------------------------------
__global__ __launch_bounds__(256) void qkv_proj_kernel(
    const float* __restrict__ x, const int* __restrict__ pos,
    const float* __restrict__ q_proj, const float* __restrict__ kv_proj,
    const float* __restrict__ qns, const float* __restrict__ kns,
    unsigned short* __restrict__ qbuf, unsigned short* __restrict__ kbuf,
    unsigned short* __restrict__ vbuf)
{
  const int t0  = blockIdx.x * 16;
  const int p   = blockIdx.y;            // 0..15 Q heads, 16..23 K heads, 24..31 V heads
  const int b   = blockIdx.z;
  const int tid = threadIdx.x;
  const int wave = tid >> 5, lane = tid & 31;
  const int row  = lane & 15, hi = lane >> 4;

  const float* W;
  unsigned short* dst;
  int mode;
  if (p < NQH) {
    mode = 0; W = q_proj + (size_t)p * DMODEL * HDIM;
    dst = qbuf + ((size_t)(b * NQH + p) * TSEQ + t0) * HDIM;
  } else if (p < NQH + NKH) {
    int kh = p - NQH; mode = 1;
    W = kv_proj + (size_t)kh * DMODEL * HDIM;
    dst = kbuf + ((size_t)(b * NKH + kh) * TSEQ + t0) * HDIM;
  } else {
    int kh = p - NQH - NKH; mode = 2;
    W = kv_proj + (size_t)(NKH + kh) * DMODEL * HDIM;
    dst = vbuf + ((size_t)(b * NKH + kh) * TSEQ + t0) * HDIM;
  }

  __shared__ __attribute__((aligned(16))) unsigned short xs[16 * 128];
  __shared__ __attribute__((aligned(16))) unsigned short wls[128 * 136]; // [col][k], padded
  __shared__ float ytile[16 * 128];
  __shared__ float redA[256];
  __shared__ float rowinv[16];

  FragC acc;
  #pragma unroll
  for (int i = 0; i < 8; i++) acc.f[i] = 0.f;

  const float* xrowbase = x + ((size_t)b * TSEQ + t0) * DMODEL;

  for (int kc = 0; kc < DMODEL; kc += 128) {
    for (int i = tid; i < 16 * 128; i += 256) {
      int r = i >> 7, k = i & 127;
      xs[i] = f2bf(xrowbase[(size_t)r * DMODEL + kc + k]);
    }
    for (int i = tid; i < 128 * 128; i += 256) {
      int k = i >> 7, c = i & 127;                     // coalesced over c
      wls[c * 136 + k] = f2bf(W[(size_t)(kc + k) * HDIM + c]);
    }
    __syncthreads();
    #pragma unroll
    for (int k = 0; k < 128; k += 32) {
      FragAB a, bb;
      const unsigned short* ap = &xs[row * 128 + k + hi * 8];
      a.u[0] = *reinterpret_cast<const uint4*>(ap);
      a.u[1] = *reinterpret_cast<const uint4*>(ap + 16);
      const unsigned short* bp = &wls[(wave * 16 + row) * 136 + k + hi * 8];
      bb.u[0] = *reinterpret_cast<const uint4*>(bp);
      bb.u[1] = *reinterpret_cast<const uint4*>(bp + 16);
      acc.v = wmma_bf16(a, bb, acc.v);
    }
    __syncthreads();
  }

  // scatter C tile (lane=col, vgpr=row) into LDS for cross-wave epilogue
  #pragma unroll
  for (int v = 0; v < 8; v++)
    ytile[(v + hi * 8) * 128 + wave * 16 + row] = acc.f[v];
  __syncthreads();

  if (mode == 2) {                                    // V: plain bf16 store
    for (int i = tid; i < 16 * 128; i += 256)
      dst[(i >> 7) * HDIM + (i & 127)] = f2bf(ytile[i]);
  } else {                                            // Q/K: RMSNorm + RoPE
    {
      int r = tid >> 4, j = tid & 15;
      float ss = 0.f;
      for (int h = j; h < 128; h += 16) { float yv = ytile[r * 128 + h]; ss += yv * yv; }
      redA[tid] = ss;
    }
    __syncthreads();
    if (tid < 16) {
      float ss = 0.f;
      for (int j = 0; j < 16; j++) ss += redA[tid * 16 + j];
      rowinv[tid] = rsqrtf(ss * (1.0f / 128.0f) + 1e-6f);
    }
    __syncthreads();
    const float* ns = (mode == 0) ? qns : kns;
    const float qscale = (mode == 0) ? (1.0f / 128.0f) : 1.0f; // q*H^-0.5 AND /sqrt(H)
    for (int i = tid; i < 16 * 64; i += 256) {
      int r = i >> 6, h = i & 63;
      float inv = rowinv[r];
      float x1 = ytile[r * 128 + h]      * inv * (1.0f + ns[h]);
      float x2 = ytile[r * 128 + h + 64] * inv * (1.0f + ns[h + 64]);
      float pf  = (float)pos[b * TSEQ + t0 + r];
      float ts  = __powf(10000.0f, (float)h * (2.0f / 128.0f));
      float ang = pf / ts;
      float sn = sinf(ang), cs = cosf(ang);
      dst[r * HDIM + h]      = f2bf((x1 * cs - x2 * sn) * qscale);
      dst[r * HDIM + h + 64] = f2bf((x2 * cs + x1 * sn) * qscale);
    }
  }
}

// ---------------------------------------------------------------------------
// Kernel B: GQA causal flash attention, bf16 in -> bf16 O (in place over Q)
// grid (T/16, NQ, B), 128 threads (4 waves)
// Q tile staged via TDM tensor_load_to_lds; V tiles via async global->LDS.
// ---------------------------------------------------------------------------
__global__ __launch_bounds__(128) void attn_kernel(
    const unsigned short* __restrict__ qbuf,
    const unsigned short* __restrict__ kbuf,
    const unsigned short* __restrict__ vbuf,
    unsigned short* obuf)
{
  const int q0  = blockIdx.x * 16;
  const int n   = blockIdx.y;
  const int b   = blockIdx.z;
  const int kh  = n >> 1;                         // group = NQ/NK = 2
  const int tid = threadIdx.x;
  const int wave = tid >> 5, lane = tid & 31;
  const int row  = lane & 15, hi = lane >> 4;

  const unsigned short* qp = qbuf + ((size_t)(b * NQH + n) * TSEQ + q0) * HDIM;
  const unsigned short* kp = kbuf + (size_t)(b * NKH + kh) * TSEQ * HDIM;
  const unsigned short* vp = vbuf + (size_t)(b * NKH + kh) * TSEQ * HDIM;

  __shared__ __attribute__((aligned(16))) unsigned short qs[16 * 128];
  __shared__ float slds[16 * 64];
  __shared__ __attribute__((aligned(16))) unsigned short plds[16 * 64];
  __shared__ __attribute__((aligned(16))) unsigned short vraw[64 * 128];  // [key][h]
  __shared__ float redbuf[16 * 8];
  __shared__ float mrow[16], lrow[16], arow[16], mnew[16];

  // --- Q tile: one TDM descriptor (contiguous 2048 bf16), wave 0 issues ---
  if (wave == 0) {
    tdm_load_bf16(lds_offset(qs), qp,
                  /*tensor_d0=*/2048, /*tensor_d1=*/1,
                  /*tile_d0=*/2048, /*tile_d1=*/0,
                  /*stride0=*/2048, /*stride1=*/2048);
    __builtin_amdgcn_s_wait_tensorcnt(0);
  }
  if (tid < 16) { mrow[tid] = -3.0e38f; lrow[tid] = 0.f; }
  FragC o0, o1;
  #pragma unroll
  for (int i = 0; i < 8; i++) { o0.f[i] = 0.f; o1.f[i] = 0.f; }
  __syncthreads();

  const unsigned vraw_off = lds_offset(vraw);
  const int kend = q0 + 16;
  for (int kt0 = 0; kt0 < kend; kt0 += 64) {
    // --- stage V tile [64 keys][128 h] via async global->LDS (16 KB) ---
    {
      const unsigned short* vsrc = vp + (size_t)kt0 * HDIM;
      #pragma unroll
      for (int it = 0; it < 8; ++it) {
        unsigned e = (unsigned)((it * 128 + tid) * 8);       // bf16 element index
        async_copy_b128(vraw_off + e * 2u, vsrc + e);
      }
      async_wait0();
    }
    // prefetch next K tile while we compute on this one
    if (kt0 + 64 < kend) {
      const unsigned short* pf =
          kp + (size_t)(kt0 + 64 + (tid >> 1)) * HDIM + (tid & 1) * 64;
      __builtin_prefetch(pf, 0, 1);
    }
    __syncthreads();

    // --- scores: wave w computes keys kt0 + 16w .. +15 over full H ---
    FragC s;
    #pragma unroll
    for (int i = 0; i < 8; i++) s.f[i] = 0.f;
    #pragma unroll
    for (int hc = 0; hc < 128; hc += 32) {
      FragAB a, bb;
      const unsigned short* ap = &qs[row * 128 + hc + hi * 8];
      a.u[0] = *reinterpret_cast<const uint4*>(ap);
      a.u[1] = *reinterpret_cast<const uint4*>(ap + 16);
      const unsigned short* bp = kp + (size_t)(kt0 + wave * 16 + row) * HDIM + hc + hi * 8;
      bb.u[0] = *reinterpret_cast<const uint4*>(bp);
      bb.u[1] = *reinterpret_cast<const uint4*>(bp + 16);
      s.v = wmma_bf16(a, bb, s.v);
    }
    #pragma unroll
    for (int v = 0; v < 8; v++)
      slds[(v + hi * 8) * 64 + wave * 16 + row] = s.f[v];
    __syncthreads();

    // --- causal mask + running row max ---
    {
      int r = tid >> 3, j0 = tid & 7;
      float mx = -3.0e38f;
      for (int j = j0; j < 64; j += 8) {
        float sv = slds[r * 64 + j];
        if (kt0 + j > q0 + r) { sv = -2.3819763e38f; slds[r * 64 + j] = sv; }
        mx = fmaxf(mx, sv);
      }
      redbuf[r * 8 + j0] = mx;
    }
    __syncthreads();
    if (tid < 16) {
      float mx = mrow[tid];
      for (int j = 0; j < 8; j++) mx = fmaxf(mx, redbuf[tid * 8 + j]);
      mnew[tid] = mx;
      arow[tid] = __expf(mrow[tid] - mx);
      mrow[tid] = mx;
    }
    __syncthreads();

    // --- P = exp(S - m), bf16, + partial row sums ---
    {
      int r = tid >> 3, j0 = tid & 7;
      float mx = mnew[r], ssum = 0.f;
      for (int j = j0; j < 64; j += 8) {
        float pv = __expf(slds[r * 64 + j] - mx);
        plds[r * 64 + j] = f2bf(pv);
        ssum += pv;
      }
      redbuf[r * 8 + j0] = ssum;
    }
    __syncthreads();
    if (tid < 16) {
      float ssum = 0.f;
      for (int j = 0; j < 8; j++) ssum += redbuf[tid * 8 + j];
      lrow[tid] = lrow[tid] * arow[tid] + ssum;
    }

    // --- rescale O, then O += P * V  (wave w owns H-cols 32w..32w+31) ---
    #pragma unroll
    for (int v = 0; v < 8; v++) {
      float al = arow[v + hi * 8];
      o0.f[v] *= al; o1.f[v] *= al;
    }
    #pragma unroll
    for (int kc = 0; kc < 64; kc += 32) {
      FragAB a;
      const unsigned short* ap = &plds[row * 64 + kc + hi * 8];
      a.u[0] = *reinterpret_cast<const uint4*>(ap);
      a.u[1] = *reinterpret_cast<const uint4*>(ap + 16);
      #pragma unroll
      for (int ct = 0; ct < 2; ct++) {
        int col = wave * 32 + ct * 16 + row;           // H channel (B-operand lane)
        FragAB bb;
        #pragma unroll
        for (int j = 0; j < 16; j++)                   // gather key-rows from vraw
          bb.s[j] = vraw[(kc + hi * 8 + (j & 7) + (j >> 3) * 16) * 128 + col];
        if (ct == 0) o0.v = wmma_bf16(a, bb, o0.v);
        else         o1.v = wmma_bf16(a, bb, o1.v);
      }
    }
    __syncthreads();
  }

  unsigned short* op = obuf + ((size_t)(b * NQH + n) * TSEQ + q0) * HDIM;
  #pragma unroll
  for (int v = 0; v < 8; v++) {
    int r = v + hi * 8;
    float inv = 1.0f / lrow[r];
    op[r * HDIM + wave * 32 + row]      = f2bf(o0.f[v] * inv);
    op[r * HDIM + wave * 32 + 16 + row] = f2bf(o1.f[v] * inv);
  }
}

// ---------------------------------------------------------------------------
// Kernel C: output projection  out[b,t,d] = sum_{n,h} O[b,n,t,h] * Wo[n,h,d]
// grid (T/16, D/128, B), 256 threads (8 waves)
// ---------------------------------------------------------------------------
__global__ __launch_bounds__(256) void oproj_kernel(
    const unsigned short* __restrict__ obuf,
    const float* __restrict__ o_proj,
    float* __restrict__ out)
{
  const int t0  = blockIdx.x * 16;
  const int d0  = blockIdx.y * 128;
  const int b   = blockIdx.z;
  const int tid = threadIdx.x;
  const int wave = tid >> 5, lane = tid & 31;
  const int row  = lane & 15, hi = lane >> 4;

  __shared__ __attribute__((aligned(16))) unsigned short wls[128 * 136];

  FragC acc;
  #pragma unroll
  for (int i = 0; i < 8; i++) acc.f[i] = 0.f;

  for (int kc = 0; kc < NQH * HDIM; kc += 128) {
    for (int i = tid; i < 128 * 128; i += 256) {
      int k = i >> 7, c = i & 127;
      wls[c * 136 + k] = f2bf(o_proj[(size_t)(kc + k) * DMODEL + d0 + c]);
    }
    __syncthreads();
    #pragma unroll
    for (int k = 0; k < 128; k += 32) {
      int kk = kc + k;
      int n = kk >> 7, h = kk & 127;                  // 32-chunks never cross heads
      FragAB a, bb;
      const unsigned short* ap =
          obuf + ((size_t)(b * NQH + n) * TSEQ + t0 + row) * HDIM + h + hi * 8;
      a.u[0] = *reinterpret_cast<const uint4*>(ap);
      a.u[1] = *reinterpret_cast<const uint4*>(ap + 16);
      const unsigned short* bp = &wls[(wave * 16 + row) * 136 + k + hi * 8];
      bb.u[0] = *reinterpret_cast<const uint4*>(bp);
      bb.u[1] = *reinterpret_cast<const uint4*>(bp + 16);
      acc.v = wmma_bf16(a, bb, acc.v);
    }
    __syncthreads();
  }
  #pragma unroll
  for (int v = 0; v < 8; v++) {
    int r = v + hi * 8;
    out[((size_t)b * TSEQ + t0 + r) * DMODEL + d0 + wave * 16 + row] = acc.f[v];
  }
}

// ---------------------------------------------------------------------------
extern "C" void kernel_launch(void* const* d_in, const int* in_sizes, int n_in,
                              void* d_out, int out_size, void* d_ws, size_t ws_size,
                              hipStream_t stream) {
  const float* x       = (const float*)d_in[0];
  /* d_in[1] = mask (bool): causal, applied analytically */
  const int*   pos     = (const int*)d_in[2];
  const float* q_proj  = (const float*)d_in[3];
  const float* kv_proj = (const float*)d_in[4];
  const float* o_proj  = (const float*)d_in[5];
  const float* qns     = (const float*)d_in[6];
  const float* kns     = (const float*)d_in[7];
  float* out = (float*)d_out;

  // workspace: Q (reused in-place as O) | K | V, all bf16 — 67 MB total, L2-resident
  unsigned short* qbuf = (unsigned short*)d_ws;
  unsigned short* kbuf = qbuf + (size_t)BATCH * NQH * TSEQ * HDIM;
  unsigned short* vbuf = kbuf + (size_t)BATCH * NKH * TSEQ * HDIM;

  dim3 gA(TSEQ / 16, NQH + 2 * NKH, BATCH);
  qkv_proj_kernel<<<gA, 256, 0, stream>>>(x, pos, q_proj, kv_proj, qns, kns,
                                          qbuf, kbuf, vbuf);
  dim3 gB(TSEQ / 16, NQH, BATCH);
  attn_kernel<<<gB, 128, 0, stream>>>(qbuf, kbuf, vbuf, qbuf);
  dim3 gC(TSEQ / 16, DMODEL / 128, BATCH);
  oproj_kernel<<<gC, 256, 0, stream>>>(qbuf, o_proj, out);
}
